// DocREModel_11038065951138
// MI455X (gfx1250) — matrix operand order repaired and checked by reading, not compiled, not run on hardware
//
#include <hip/hip_runtime.h>
#include <hip/hip_bf16.h>
#include <math.h>

// ---------------------------------------------------------------------------
// Problem constants (from reference)
// ---------------------------------------------------------------------------
#define CB   4          // batch
#define CL   1024       // seq len
#define CH   768        // hidden
#define CE   32         // entities
#define CM   8          // mentions (GRU time steps)
#define CP   992        // pairs per batch
#define CEMB 768
#define CBLK 64
#define CNL  97
#define CG3  (3*CH)     // 2304
#define CN   (CB*CE)    // 128 sequences
#define CNT  (CN*CM)    // 1024 rows of x
#define CKH  (2*CH)     // 1536
#define CNP  (CB*CP)    // 3968 pairs
#define CKB  (CEMB*CBLK)// 49152
#define KSPLIT 8        // K-split for bilinear classifier

typedef __attribute__((ext_vector_type(16))) __bf16 v16bf;
typedef __attribute__((ext_vector_type(8)))  float  v8f;
typedef __attribute__((ext_vector_type(4)))  unsigned u32x4;
typedef __attribute__((ext_vector_type(4)))  int      i32x4;

typedef __attribute__((address_space(1))) i32x4 gi32x4;
typedef __attribute__((address_space(3))) i32x4 li32x4;

union Frag16 {
    unsigned u[8];
    __bf16   e[16];
    v16bf    v;
};

#if defined(__has_builtin)
#if __has_builtin(__builtin_amdgcn_global_load_async_to_lds_b128) && \
    __has_builtin(__builtin_amdgcn_s_wait_asynccnt)
#define USE_ASYNC_LDS 1
#endif
#endif

// ---------------------------------------------------------------------------
// Elementwise helpers
// ---------------------------------------------------------------------------
__global__ void f32_to_bf16_kernel(const float* __restrict__ in,
                                   __bf16* __restrict__ out, int n) {
    int i = blockIdx.x * blockDim.x + threadIdx.x;
    if (i < n) out[i] = (__bf16)in[i];
}

__global__ void zero_f32_kernel(float* __restrict__ p, int n) {
    int i = blockIdx.x * blockDim.x + threadIdx.x;
    if (i < n) p[i] = 0.0f;
}

__global__ void zero_u16_kernel(unsigned short* __restrict__ p, int n) {
    int i = blockIdx.x * blockDim.x + threadIdx.x;
    if (i < n) p[i] = 0;
}

// Gather mention windows into x (bf16), row = (b*E + e)*M + m
__global__ void gather_x_kernel(const float* __restrict__ seq,
                                const int* __restrict__ mpos,
                                __bf16* __restrict__ xout) {
    int idx = blockIdx.x * blockDim.x + threadIdx.x;
    if (idx >= CNT * CH) return;
    int nrow = idx / CH;
    int j    = idx % CH;
    int b    = nrow / (CE * CM);
    int rem  = nrow % (CE * CM);
    int pos  = mpos[b * CE * CM + rem] + 1; // OFFSET = 1
    xout[idx] = (__bf16)seq[((size_t)(b * CL + pos)) * CH + j];
}

// One GRU time step; accumulates time-mean into ent_acc
__global__ void gru_step_kernel(const float* __restrict__ gx,
                                const float* __restrict__ gh,
                                float* __restrict__ h,
                                __bf16* __restrict__ h_bf,
                                float* __restrict__ ent_acc,
                                int t, int colOfs) {
    int idx = blockIdx.x * blockDim.x + threadIdx.x;
    if (idx >= CN * CH) return;
    int n = idx / CH;
    int j = idx % CH;
    const float* gxr = gx + ((size_t)(n * CM + t)) * CG3;
    const float* ghr = gh + (size_t)n * CG3;
    float xr = gxr[j], xz = gxr[CH + j], xn = gxr[2 * CH + j];
    float hr = ghr[j], hz = ghr[CH + j], hn = ghr[2 * CH + j];
    float hprev = h[idx];
    float r = 1.0f / (1.0f + __expf(-(xr + hr)));
    float z = 1.0f / (1.0f + __expf(-(xz + hz)));
    float nn = tanhf(xn + r * hn);
    float hnew = (1.0f - z) * nn + z * hprev;
    h[idx]    = hnew;
    h_bf[idx] = (__bf16)hnew;
    ent_acc[(size_t)n * CKH + colOfs + j] += hnew;
}

__global__ void ent_finalize_kernel(const float* __restrict__ acc,
                                    __bf16* __restrict__ ent_bf, int n) {
    int i = blockIdx.x * blockDim.x + threadIdx.x;
    if (i < n) ent_bf[i] = (__bf16)(acc[i] * 0.125f);
}

__global__ void gather_pairs_kernel(const __bf16* __restrict__ entbf,
                                    const int* __restrict__ hts,
                                    __bf16* __restrict__ hsb,
                                    __bf16* __restrict__ tsb) {
    int idx = blockIdx.x * blockDim.x + threadIdx.x;
    if (idx >= CNP * CKH) return;
    int p = idx / CKH;
    int j = idx % CKH;
    int b = p / CP;
    int hi = hts[p * 2 + 0];
    int ti = hts[p * 2 + 1];
    hsb[idx] = entbf[((size_t)(b * CE + hi)) * CKH + j];
    tsb[idx] = entbf[((size_t)(b * CE + ti)) * CKH + j];
}

// ---------------------------------------------------------------------------
// Shared GEMM helpers
// ---------------------------------------------------------------------------
__device__ __forceinline__ void stage_panel(const __bf16* src, unsigned short* ldst) {
#ifdef USE_ASYNC_LDS
    __builtin_amdgcn_global_load_async_to_lds_b128((gi32x4*)src, (li32x4*)ldst, 0, 0);
#else
    *(u32x4*)ldst = *(const u32x4*)src;
#endif
}

__device__ __forceinline__ void wait_stage() {
#ifdef USE_ASYNC_LDS
    __builtin_amdgcn_s_wait_asynccnt(0);
#endif
    __syncthreads();
}

__device__ __forceinline__ void load_afrag(const __bf16* Ap, int kk, int kbA, Frag16* a) {
#pragma unroll
    for (int i = 0; i < 4; ++i) {
        a->u[i]     = *(const unsigned*)(Ap + kk + kbA + 2 * i);
        a->u[4 + i] = *(const unsigned*)(Ap + kk + 16 + kbA + 2 * i);
    }
}

__device__ __forceinline__ void wmma4(const Frag16& a, const unsigned short* pan,
                                      int lane, int kbB,
                                      v8f& c0, v8f& c1, v8f& c2, v8f& c3) {
#pragma unroll
    for (int s = 0; s < 4; ++s) {
        const unsigned short* bp = pan + ((lane & 15) + s * 16) * 32 + kbB;
        Frag16 b;
        *(u32x4*)&b.u[0] = *(const u32x4*)bp;
        *(u32x4*)&b.u[4] = *(const u32x4*)(bp + 16);
        v8f& c = (s == 0) ? c0 : (s == 1) ? c1 : (s == 2) ? c2 : c3;
        c = __builtin_amdgcn_wmma_f32_16x16x32_bf16(false, a.v, false, b.v,
                                                    (short)0, c, false, false);
    }
}

// ---------------------------------------------------------------------------
// 128x64 block-tile bf16 WMMA GEMM: C[m,n] = sum_k A[m,k]*W[n,k] (+bias[n])
//   8 waves/block; wave w owns rows [w*16,w*16+16), 4 accumulators over 64
//   cols. B panel (64x32, 4KB) double-buffered in LDS via async global->LDS,
//   software pipelined: the NEXT panel's async copy and NEXT A fragment are
//   issued before each 4-WMMA block, hiding load latency behind matrix ops.
//   Requires: Mr % 128 == 0, Nc % 64 == 0, K % 64 == 0 (true for all uses).
//   EPI: 0 = f32 store, 1 = relu->bf16, 2 = tanh->bf16
// ---------------------------------------------------------------------------
template <int EPI>
__global__ __launch_bounds__(256) void wmma_gemm_kernel(
    const __bf16* __restrict__ A, int lda,
    const __bf16* __restrict__ W, int ldw,
    const float* __restrict__ bias,
    void* __restrict__ Cout, int ldc,
    int Mr, int Nc, int K) {
    __shared__ __align__(16) unsigned short lpan[2][64 * 32]; // 2 x 4KB

    int tid  = threadIdx.x;
    int lane = tid & 31;
    int wv   = tid >> 5;
    int tilesN = Nc >> 6;
    int bm = blockIdx.x / tilesN;
    int bn = blockIdx.x % tilesN;

    int row = bm * 128 + wv * 16 + (lane & 15);
    int kbA = (lane >> 4) * 8;
    int kbB = (lane >> 4) * 8;
    const __bf16* Ap = A + (size_t)row * lda;

    // cooperative B-panel loader: thread -> (col, 8-element chunk)
    int lcol   = tid >> 2;        // 0..63
    int lchunk = (tid & 3) * 8;   // 0,8,16,24 (elements)
    const __bf16* Wl = W + (size_t)(bn * 64 + lcol) * ldw + lchunk;
    unsigned short* lput0 = &lpan[0][lcol * 32 + lchunk];
    unsigned short* lput1 = &lpan[1][lcol * 32 + lchunk];

    v8f acc0 = {}, acc1 = {}, acc2 = {}, acc3 = {};
    Frag16 aP, aQ;

    // prologue: stage panel(0) -> buf0, load A(0)
    stage_panel(Wl, lput0);
    load_afrag(Ap, 0, kbA, &aP);
    wait_stage();

#pragma unroll 1
    for (int kk = 0; kk < K; kk += 64) {
        // overlap: stage panel(kk+32) -> buf1, load A(kk+32), behind WMMAs on buf0
        stage_panel(Wl + kk + 32, lput1);
        load_afrag(Ap, kk + 32, kbA, &aQ);
        wmma4(aP, &lpan[0][0], lane, kbB, acc0, acc1, acc2, acc3);
        wait_stage();

        // overlap: stage panel(kk+64) -> buf0, load A(kk+64), behind WMMAs on buf1
        if (kk + 64 < K) {
            stage_panel(Wl + kk + 64, lput0);
            load_afrag(Ap, kk + 64, kbA, &aP);
        }
        wmma4(aQ, &lpan[1][0], lane, kbB, acc0, acc1, acc2, acc3);
        wait_stage();
    }

    // epilogue
#pragma unroll
    for (int s = 0; s < 4; ++s) {
        v8f acc = (s == 0) ? acc0 : (s == 1) ? acc1 : (s == 2) ? acc2 : acc3;
        int col = bn * 64 + s * 16 + (lane & 15);
        float bv = bias ? bias[col] : 0.0f;
#pragma unroll
        for (int r = 0; r < 8; ++r) {
            int orow = bm * 128 + wv * 16 + r + ((lane >> 4) * 8);
            float v = acc[r] + bv;
            if (EPI == 0)
                ((float*)Cout)[(size_t)orow * ldc + col] = v;
            else if (EPI == 1)
                ((__bf16*)Cout)[(size_t)orow * ldc + col] = (__bf16)fmaxf(v, 0.0f);
            else
                ((__bf16*)Cout)[(size_t)orow * ldc + col] = (__bf16)tanhf(v);
        }
    }
}

// ---------------------------------------------------------------------------
// Fused block-bilinear + classifier, K-split into partial sums:
//   logits[p,l] = sum_{k,b,c} hs[p,k*64+b] * ts[p,k*64+c] * wb[l,k*4096+b*64+c]
// GEMM view: M=3968, N=97(->112), K=49152. Within an aligned K=32 chunk, k and
// b are constant: each lane synthesizes its bf16 A fragment from one hs scalar
// and a 32-element ts run (no 780 MB `bl` intermediate). Each wave owns one M
// tile, ALL 7 N tiles and 1/8 of K. Fragments are double-buffered: chunk it+1
// is fetched/synthesized while chunk it's 7 WMMAs run.
// ---------------------------------------------------------------------------
__device__ __forceinline__ void bil_load_frags(
    int it, const __bf16* hrow, const __bf16* trow,
    const __bf16* const* wrow, int kbA, int kbB,
    Frag16* a, Frag16* b) {
    int K0   = it * 32;
    int kblk = K0 >> 12;
    int bidx = (K0 >> 6) & 63;
    int c0   = K0 & 63;
    float hv = (float)hrow[kblk * CBLK + bidx];
    const __bf16* tbase = trow + kblk * CBLK + c0;
#pragma unroll
    for (int i = 0; i < 8; ++i) {
        int off = kbA + 2 * i;
        a->e[2 * i]     = (__bf16)(hv * (float)tbase[off]);
        a->e[2 * i + 1] = (__bf16)(hv * (float)tbase[off + 1]);
    }
#pragma unroll
    for (int t = 0; t < 7; ++t) {
#pragma unroll
        for (int i = 0; i < 8; ++i)
            b[t].u[i] = *(const unsigned*)(wrow[t] + K0 + kbB + 2 * i);
    }
}

__global__ __launch_bounds__(256) void bilinear_wmma_kernel(
    const __bf16* __restrict__ hsp,
    const __bf16* __restrict__ tsp,
    const __bf16* __restrict__ wbb,
    float* __restrict__ part) {
    const int tilesM = CNP / 16; // 248
    int lane = threadIdx.x & 31;
    int gw   = blockIdx.x * 8 + (threadIdx.x >> 5);
    if (gw >= tilesM * KSPLIT) return;
    int tm = gw / KSPLIT;
    int ks = gw % KSPLIT;

    int p   = tm * 16 + (lane & 15);
    int kbA = (lane >> 4) * 16;
    int kbB = (lane >> 4) * 16;
    const __bf16* hrow = hsp + (size_t)p * CEMB;
    const __bf16* trow = tsp + (size_t)p * CEMB;

    const __bf16* wrow[7];
#pragma unroll
    for (int t = 0; t < 7; ++t) {
        int nc = t * 16 + (lane & 15);
        if (nc >= CNL) nc = CNL - 1; // clamp pad columns
        wrow[t] = wbb + (size_t)nc * CKB;
    }

    v8f acc[7] = {};
    const int it0 = ks * (CKB / 32 / KSPLIT); // 192 chunks per slice (even)
    const int it1 = it0 + (CKB / 32 / KSPLIT);

    Frag16 aP, aQ;
    Frag16 bP[7], bQ[7];
    bil_load_frags(it0, hrow, trow, wrow, kbA, kbB, &aP, bP);

#pragma unroll 1
    for (int it = it0; it < it1; it += 2) {
        // fetch chunk it+1 behind the 7 WMMAs of chunk it
        bil_load_frags(it + 1, hrow, trow, wrow, kbA, kbB, &aQ, bQ);
#pragma unroll
        for (int t = 0; t < 7; ++t)
            acc[t] = __builtin_amdgcn_wmma_f32_16x16x32_bf16(
                false, aP.v, false, bP[t].v, (short)0, acc[t], false, false);

        // fetch chunk it+2 behind the 7 WMMAs of chunk it+1
        if (it + 2 < it1)
            bil_load_frags(it + 2, hrow, trow, wrow, kbA, kbB, &aP, bP);
#pragma unroll
        for (int t = 0; t < 7; ++t)
            acc[t] = __builtin_amdgcn_wmma_f32_16x16x32_bf16(
                false, aQ.v, false, bQ[t].v, (short)0, acc[t], false, false);
    }

    float* pslice = part + (size_t)ks * CNP * CNL;
#pragma unroll
    for (int t = 0; t < 7; ++t) {
        int ncol = t * 16 + (lane & 15);
        if (ncol < CNL) {
#pragma unroll
            for (int r = 0; r < 8; ++r) {
                int orow = tm * 16 + r + ((lane >> 4) * 8);
                pslice[(size_t)orow * CNL + ncol] = acc[t][r];
            }
        }
    }
}

__global__ void bilinear_reduce_kernel(const float* __restrict__ part,
                                       const float* __restrict__ bb,
                                       float* __restrict__ out) {
    int i = blockIdx.x * blockDim.x + threadIdx.x;
    if (i >= CNP * CNL) return;
    float s = bb[i % CNL];
#pragma unroll
    for (int k = 0; k < KSPLIT; ++k) s += part[(size_t)k * CNP * CNL + i];
    out[i] = s;
}

// ---------------------------------------------------------------------------
// Host-side orchestration
// ---------------------------------------------------------------------------
static inline size_t alignup(size_t x) { return (x + 255) & ~(size_t)255; }

extern "C" void kernel_launch(void* const* d_in, const int* in_sizes, int n_in,
                              void* d_out, int out_size, void* d_ws, size_t ws_size,
                              hipStream_t stream) {
    (void)in_sizes; (void)n_in; (void)out_size; (void)ws_size;

    const float* seq    = (const float*)d_in[0];
    const int*   mpos   = (const int*)d_in[1];
    const int*   hts    = (const int*)d_in[2];
    const float* w_ih_f = (const float*)d_in[3];
    const float* w_hh_f = (const float*)d_in[4];
    const float* b_ih_f = (const float*)d_in[5];
    const float* b_hh_f = (const float*)d_in[6];
    const float* w_ih_b = (const float*)d_in[7];
    const float* w_hh_b = (const float*)d_in[8];
    const float* b_ih_b = (const float*)d_in[9];
    const float* b_hh_b = (const float*)d_in[10];
    const float* w1     = (const float*)d_in[11];
    const float* b1     = (const float*)d_in[12];
    const float* w2     = (const float*)d_in[13];
    const float* b2     = (const float*)d_in[14];
    const float* wb     = (const float*)d_in[15];
    const float* bb     = (const float*)d_in[16];
    float* out = (float*)d_out;

    char* base = (char*)d_ws;
    size_t off = 0;
    auto take = [&](size_t bytes) { char* q = base + off; off += alignup(bytes); return q; };

    __bf16* x_bf    = (__bf16*)take((size_t)CNT * CH * 2);
    __bf16* wihf_bf = (__bf16*)take((size_t)CG3 * CH * 2);
    __bf16* whhf_bf = (__bf16*)take((size_t)CG3 * CH * 2);
    __bf16* wihb_bf = (__bf16*)take((size_t)CG3 * CH * 2);
    __bf16* whhb_bf = (__bf16*)take((size_t)CG3 * CH * 2);
    __bf16* w1_bf   = (__bf16*)take((size_t)CEMB * CKH * 2);
    __bf16* w2_bf   = (__bf16*)take((size_t)CEMB * CEMB * 2);
    __bf16* wb_bf   = (__bf16*)take((size_t)CNL * CKB * 2);
    float*  gx_f    = (float*)take((size_t)CNT * CG3 * 4);
    float*  gx_b    = (float*)take((size_t)CNT * CG3 * 4);
    float*  gh      = (float*)take((size_t)CN * CG3 * 4);
    float*  h_f     = (float*)take((size_t)CN * CH * 4);
    __bf16* h_bf    = (__bf16*)take((size_t)CN * CH * 2);
    float*  ent_acc = (float*)take((size_t)CN * CKH * 4);
    __bf16* ent_bf  = (__bf16*)take((size_t)CN * CKH * 2);
    __bf16* hs_bf   = (__bf16*)take((size_t)CNP * CKH * 2);
    __bf16* ts_bf   = (__bf16*)take((size_t)CNP * CKH * 2);
    __bf16* hs1_bf  = (__bf16*)take((size_t)CNP * CEMB * 2);
    __bf16* ts1_bf  = (__bf16*)take((size_t)CNP * CEMB * 2);
    __bf16* hsp_bf  = (__bf16*)take((size_t)CNP * CEMB * 2);
    __bf16* tsp_bf  = (__bf16*)take((size_t)CNP * CEMB * 2);
    float*  part    = (float*)take((size_t)KSPLIT * CNP * CNL * 4);

    auto conv = [&](const float* src, __bf16* dst, int n) {
        f32_to_bf16_kernel<<<(n + 255) / 256, 256, 0, stream>>>(src, dst, n);
    };
    auto gemm = [&](int epi, const __bf16* A, int lda, const __bf16* W, int ldw,
                    const float* bias, void* C, int ldc, int Mr, int Nc, int K) {
        int blocks = (Mr / 128) * (Nc / 64);
        if (epi == 0)
            wmma_gemm_kernel<0><<<blocks, 256, 0, stream>>>(A, lda, W, ldw, bias, C, ldc, Mr, Nc, K);
        else if (epi == 1)
            wmma_gemm_kernel<1><<<blocks, 256, 0, stream>>>(A, lda, W, ldw, bias, C, ldc, Mr, Nc, K);
        else
            wmma_gemm_kernel<2><<<blocks, 256, 0, stream>>>(A, lda, W, ldw, bias, C, ldc, Mr, Nc, K);
    };

    // 1) Weight conversions to bf16
    conv(w_ih_f, wihf_bf, CG3 * CH);
    conv(w_hh_f, whhf_bf, CG3 * CH);
    conv(w_ih_b, wihb_bf, CG3 * CH);
    conv(w_hh_b, whhb_bf, CG3 * CH);
    conv(w1, w1_bf, CEMB * CKH);
    conv(w2, w2_bf, CEMB * CEMB);
    conv(wb, wb_bf, CNL * CKB);

    // 2) Mention gather -> x (bf16)
    {
        int n = CNT * CH;
        gather_x_kernel<<<(n + 255) / 256, 256, 0, stream>>>(seq, mpos, x_bf);
    }

    // 3) GRU input transforms (both directions)
    gemm(0, x_bf, CH, wihf_bf, CH, b_ih_f, gx_f, CG3, CNT, CG3, CH);
    gemm(0, x_bf, CH, wihb_bf, CH, b_ih_b, gx_b, CG3, CNT, CG3, CH);

    // 4) Zero accumulators
    {
        int n = CN * CKH;
        zero_f32_kernel<<<(n + 255) / 256, 256, 0, stream>>>(ent_acc, n);
    }
    auto zero_h = [&]() {
        int n = CN * CH;
        zero_f32_kernel<<<(n + 255) / 256, 256, 0, stream>>>(h_f, n);
        zero_u16_kernel<<<(n + 255) / 256, 256, 0, stream>>>((unsigned short*)h_bf, n);
    };

    const int stepBlocks = (CN * CH + 255) / 256;

    // 5) Forward GRU scan
    zero_h();
    for (int t = 0; t < CM; ++t) {
        gemm(0, h_bf, CH, whhf_bf, CH, b_hh_f, gh, CG3, CN, CG3, CH);
        gru_step_kernel<<<stepBlocks, 256, 0, stream>>>(gx_f, gh, h_f, h_bf, ent_acc, t, 0);
    }

    // 6) Backward GRU scan
    zero_h();
    for (int t = CM - 1; t >= 0; --t) {
        gemm(0, h_bf, CH, whhb_bf, CH, b_hh_b, gh, CG3, CN, CG3, CH);
        gru_step_kernel<<<stepBlocks, 256, 0, stream>>>(gx_b, gh, h_f, h_bf, ent_acc, t, CH);
    }

    // 7) ent = acc / 8 -> bf16
    {
        int n = CN * CKH;
        ent_finalize_kernel<<<(n + 255) / 256, 256, 0, stream>>>(ent_acc, ent_bf, n);
    }

    // 8) Pair gathers
    {
        int n = CNP * CKH;
        gather_pairs_kernel<<<(n + 255) / 256, 256, 0, stream>>>(ent_bf, hts, hs_bf, ts_bf);
    }

    // 9) Projections
    gemm(1, hs_bf, CKH, w1_bf, CKH, b1, hs1_bf, CEMB, CNP, CEMB, CKH);
    gemm(2, hs1_bf, CEMB, w2_bf, CEMB, b2, hsp_bf, CEMB, CNP, CEMB, CEMB);
    gemm(1, ts_bf, CKH, w1_bf, CKH, b1, ts1_bf, CEMB, CNP, CEMB, CKH);
    gemm(2, ts1_bf, CEMB, w2_bf, CEMB, b2, tsp_bf, CEMB, CNP, CEMB, CEMB);

    // 10) Fused bilinear classifier (K-split partials) + deterministic reduce
    {
        int waves  = (CNP / 16) * KSPLIT; // 1984
        int blocks = (waves + 7) / 8;
        bilinear_wmma_kernel<<<blocks, 256, 0, stream>>>(hsp_bf, tsp_bf, wb_bf, part);
        int n = CNP * CNL;
        bilinear_reduce_kernel<<<(n + 255) / 256, 256, 0, stream>>>(part, bb, out);
    }
}
